// AdaptiveFourierFeatures_58944131170443
// MI455X (gfx1250) — compile-verified, matrix operand order checked by reading, not run on hardware
//
#include <hip/hip_runtime.h>
#include <hip/hip_bf16.h>

typedef unsigned short u16;
typedef unsigned int   u32;
typedef __attribute__((ext_vector_type(4)))  u32    u32x4;
typedef __attribute__((ext_vector_type(16))) __bf16 v16bf;
typedef __attribute__((ext_vector_type(8)))  float  v8f;

#define BB 8
#define SS 2048
#define DD 64
#define FF 16
#define EE 32
#define HH 4
#define CI 2112              // D + 2*D*F
#define KT 66                // CI / 32
#define WELEMS (KT*4*32*16)  // 135168 == 64*2112

union FragBF { v16bf v; u32x4 q[2]; };

__device__ __forceinline__ u16 f2bf(float f) {
    u32 u = __float_as_uint(f);
    u32 r = u + 0x7FFFu + ((u >> 16) & 1u);   // round-to-nearest-even
    return (u16)(r >> 16);
}

// ---------- prep: freqs = freq_matrix * freq_scale ----------
__global__ void k_prep_freqs(const float* __restrict__ fm, const float* __restrict__ fs,
                             float* __restrict__ freqs) {
    int i = blockIdx.x * 256 + threadIdx.x;
    if (i < DD * FF) freqs[i] = fm[i] * fs[i];
}

// ---------- prep: fold query path + rank-1 key path into wq[4][64], bqh[4] ----------
__global__ void k_prep_wq(const float* __restrict__ Wq_in, const float* __restrict__ bq_in,
                          const float* __restrict__ Wk_in, const float* __restrict__ bk_in,
                          const float* __restrict__ Wq_attn, const float* __restrict__ bq_attn,
                          const float* __restrict__ Wk_attn, const float* __restrict__ bk_attn,
                          float* __restrict__ wq, float* __restrict__ bqh) {
    int t = threadIdx.x;             // 256 threads: (h, dd)
    int h = t >> 6, dd = t & 63;
    const float rs = 0.35355339059327373f;   // 1/sqrt(head_dim=8)
    float acc = 0.f, bacc = 0.f;
    for (int e = 0; e < 8; ++e) {
        int ep = h * 8 + e;
        float u = 0.f;
        for (int j = 0; j < EE; ++j) u += Wk_attn[ep * EE + j] * Wk_in[j];
        float wc = 0.f, bc = bq_attn[ep];
        for (int j = 0; j < EE; ++j) {
            float wa = Wq_attn[ep * EE + j];
            wc += wa * Wq_in[j * DD + dd];
            bc += wa * bq_in[j];
        }
        acc  += u * wc;
        bacc += u * bc;
    }
    wq[h * DD + dd] = rs * acc;
    if (dd == 0) bqh[h] = rs * bacc;   // (cancels in softmax, kept for completeness)
}

// ---------- prep: swizzle Wg/Wp into bf16 per-lane-contiguous WMMA B fragments ----------
// layout: [kt][nt][lane][i], element = W[n][k], n = nt*16 + (lane&15),
// k = kt*32 + koff(lane, i), koff = i + 8*(i>=8) + 8*(lane>=16)
__global__ void k_swz(const float* __restrict__ Wg, const float* __restrict__ Wp,
                      u16* __restrict__ wgB, u16* __restrict__ wpB) {
    int idx = blockIdx.x * 256 + threadIdx.x;
    if (idx >= 2 * WELEMS) return;
    int mat = idx >= WELEMS;
    int r = mat ? idx - WELEMS : idx;
    int i = r & 15, lane = (r >> 4) & 31, nt = (r >> 9) & 3, kt = r >> 11;
    int k = kt * 32 + i + ((i >= 8) ? 8 : 0) + ((lane >= 16) ? 8 : 0);
    int n = nt * 16 + (lane & 15);
    const float* W = mat ? Wp : Wg;
    (mat ? wpB : wgB)[r] = f2bf(W[n * CI + k]);
}

// ---------- main fused kernel: one 16-token tile per 64-thread (2-wave) block ----------
__launch_bounds__(64)
__global__ void k_main(const float* __restrict__ x, const float* __restrict__ freqs,
                       const float* __restrict__ phase, const float* __restrict__ wq,
                       const float* __restrict__ bqh, const u16* __restrict__ wgB,
                       const u16* __restrict__ wpB, const float* __restrict__ bg,
                       const float* __restrict__ bp, float* __restrict__ out) {
    __shared__ __align__(16) u16 feat[16 * CI];   // bf16 ci tile [row][k]
    __shared__ float xs[16 * DD];
    __shared__ float fr[DD * FF];
    __shared__ float phs[DD * FF];
    __shared__ float wqs[HH * DD];
    __shared__ float bqs[HH];
    __shared__ float qu[16 * HH];
    __shared__ float Cred[2 * 16 * DD];           // [mat][row][col] f32 reduction

    const int tid = threadIdx.x;
    const int b  = blockIdx.x >> 7;
    const int s0 = (blockIdx.x & 127) * 16;

    // ---- P0: stage constants, x tile (fp32 + bf16 into feat[:, :64]) ----
    for (int i = tid; i < DD * FF; i += 64) { fr[i] = freqs[i]; phs[i] = phase[i]; }
    for (int i = tid; i < HH * DD; i += 64) wqs[i] = wq[i];
    if (tid < HH) bqs[tid] = bqh[tid];
    for (int i = tid; i < 16 * DD; i += 64) {
        int row = i >> 6, col = i & 63;
        float v = x[((size_t)(b * SS + s0 + row)) * DD + col];
        xs[i] = v;
        feat[row * CI + col] = f2bf(v);
    }
    for (int i = tid; i < 2 * 16 * DD; i += 64) Cred[i] = 0.f;
    __syncthreads();

    // ---- P1: qu[row][h] = x_row . wq_h + bq_h  (attention path collapsed) ----
    {
        int row = tid >> 2, h = tid & 3;
        float acc = bqs[h];
        for (int dd2 = 0; dd2 < DD; ++dd2) acc += xs[row * DD + dd2] * wqs[h * DD + dd2];
        qu[tid] = acc;
    }
    __syncthreads();

    // ---- P2: fourier features: mean-over-head softmax * sin/cos -> bf16 LDS ----
    const float TWO_PI = 6.283185307179586f;
    for (int it = 0; it < 16; ++it) {
        int t = tid + it * 64;
        int row = t >> 6, d = t & 63;
        float qh[4] = { qu[row * 4 + 0], qu[row * 4 + 1], qu[row * 4 + 2], qu[row * 4 + 3] };
        float frd[16], w[16];
        #pragma unroll
        for (int f = 0; f < 16; ++f) { frd[f] = fr[d * FF + f]; w[f] = 0.f; }
        #pragma unroll
        for (int h = 0; h < 4; ++h) {
            float a[16], m = -3.4e38f;
            #pragma unroll
            for (int f = 0; f < 16; ++f) { a[f] = frd[f] * qh[h]; m = fmaxf(m, a[f]); }
            float ssum = 0.f;
            #pragma unroll
            for (int f = 0; f < 16; ++f) { a[f] = __expf(a[f] - m); ssum += a[f]; }
            float inv = 0.25f / ssum;
            #pragma unroll
            for (int f = 0; f < 16; ++f) w[f] += a[f] * inv;
        }
        float tm = (float)(s0 + row) * (1.0f / (float)(SS - 1));
        u16 sb[16], cb[16];
        #pragma unroll
        for (int f = 0; f < 16; ++f) {
            float ang = TWO_PI * tm * frd[f] + phs[d * FF + f];
            float sv, cv;
            __sincosf(ang, &sv, &cv);
            sb[f] = f2bf(w[f] * sv);
            cb[f] = f2bf(w[f] * cv);
        }
        u32* dst = (u32*)&feat[row * CI + DD + d * 32];
        #pragma unroll
        for (int j = 0; j < 8; ++j) {
            dst[j]     = (u32)sb[2 * j] | ((u32)sb[2 * j + 1] << 16);
            dst[8 + j] = (u32)cb[2 * j] | ((u32)cb[2 * j + 1] << 16);
        }
    }
    __syncthreads();

    // ---- P3: WMMA GEMM 16 x 2112 x 64 for gate & proj, K split across 2 waves ----
    {
        const int wave = tid >> 5, l = tid & 31;
        v8f cg[4] = {}, cp[4] = {};
        const int m = l & 15;
        const int klane = (l < 16) ? 0 : 8;
        for (int kt = wave; kt < KT; kt += 2) {
            FragBF a;
            const u32x4* pa = (const u32x4*)&feat[m * CI + kt * 32 + klane];
            a.q[0] = pa[0];           // K = base+0..7
            a.q[1] = pa[2];           // K = base+16..23 (+32 bytes)
            #pragma unroll
            for (int nt = 0; nt < 4; ++nt) {
                FragBF bgf, bpf;
                const u32x4* pg = (const u32x4*)&wgB[(((kt * 4 + nt) * 32) + l) * 16];
                bgf.q[0] = pg[0]; bgf.q[1] = pg[1];
                const u32x4* pp = (const u32x4*)&wpB[(((kt * 4 + nt) * 32) + l) * 16];
                bpf.q[0] = pp[0]; bpf.q[1] = pp[1];
                cg[nt] = __builtin_amdgcn_wmma_f32_16x16x32_bf16(
                    false, a.v, false, bgf.v, (short)0, cg[nt], false, false);
                cp[nt] = __builtin_amdgcn_wmma_f32_16x16x32_bf16(
                    false, a.v, false, bpf.v, (short)0, cp[nt], false, false);
            }
        }
        const int rowbase = (l < 16) ? 0 : 8;
        const int colb = l & 15;
        #pragma unroll
        for (int nt = 0; nt < 4; ++nt) {
            #pragma unroll
            for (int r = 0; r < 8; ++r) {
                int row = r + rowbase;
                int col = nt * 16 + colb;
                atomicAdd(&Cred[row * DD + col], cg[nt][r]);
                atomicAdd(&Cred[16 * DD + row * DD + col], cp[nt][r]);
            }
        }
    }
    __syncthreads();

    // ---- P4: epilogue: out = x + sigmoid(g) * silu(p) ----
    for (int it = 0; it < 16; ++it) {
        int i = tid + it * 64;
        int row = i >> 6, col = i & 63;
        float gpre = Cred[i] + bg[col];
        float ppre = Cred[16 * DD + i] + bp[col];
        float g  = 1.0f / (1.0f + __expf(-gpre));
        float si = ppre / (1.0f + __expf(-ppre));
        out[((size_t)(b * SS + s0 + row)) * DD + col] = xs[i] + g * si;
    }
}

extern "C" void kernel_launch(void* const* d_in, const int* in_sizes, int n_in,
                              void* d_out, int out_size, void* d_ws, size_t ws_size,
                              hipStream_t stream) {
    const float* x       = (const float*)d_in[0];
    const float* fm      = (const float*)d_in[1];
    const float* fs      = (const float*)d_in[2];
    const float* phase   = (const float*)d_in[3];
    const float* Wq_in   = (const float*)d_in[4];
    const float* bq_in   = (const float*)d_in[5];
    const float* Wk_in   = (const float*)d_in[6];
    const float* bk_in   = (const float*)d_in[7];
    const float* Wq_attn = (const float*)d_in[8];
    const float* bq_attn = (const float*)d_in[9];
    const float* Wk_attn = (const float*)d_in[10];
    const float* bk_attn = (const float*)d_in[11];
    const float* Wg      = (const float*)d_in[12];
    const float* bg      = (const float*)d_in[13];
    const float* Wp      = (const float*)d_in[14];
    const float* bp      = (const float*)d_in[15];
    float* out = (float*)d_out;

    char* ws = (char*)d_ws;
    float* freqs = (float*)ws;                 // 4096 B
    float* wq    = (float*)(ws + 4096);        // 1024 B
    float* bqh   = (float*)(ws + 5120);        // 16 B
    u16*   wgB   = (u16*)(ws + 8192);          // 270336 B
    u16*   wpB   = (u16*)(ws + 8192 + (size_t)WELEMS * 2);

    k_prep_freqs<<<4, 256, 0, stream>>>(fm, fs, freqs);
    k_prep_wq<<<1, 256, 0, stream>>>(Wq_in, bq_in, Wk_in, bk_in,
                                     Wq_attn, bq_attn, Wk_attn, bk_attn, wq, bqh);
    k_swz<<<(2 * WELEMS + 255) / 256, 256, 0, stream>>>(Wg, Wp, wgB, wpB);
    k_main<<<BB * (SS / 16), 64, 0, stream>>>(x, freqs, phase, wq, bqh,
                                              wgB, wpB, bg, bp, out);
}
